// LSTMTransformerLayer_15341623181365
// MI455X (gfx1250) — matrix operand order repaired and checked by reading, not compiled, not run on hardware
//
#include <hip/hip_runtime.h>
#include <hip/hip_bf16.h>
#include <math.h>

typedef __bf16 bf16;
typedef __attribute__((ext_vector_type(16))) __bf16 v16bf;
typedef __attribute__((ext_vector_type(8)))  __bf16 v8bf;
typedef __attribute__((ext_vector_type(8)))  float  v8f;

#define B_   16
#define T_   512
#define H_   512
#define G4H  2048
#define NTILES 128          // 4H / 16

// ---------------- workspace layout (bytes, all offsets 256B aligned) ----------------
static constexpr size_t OFF_XBF  = 0;                                   // x as bf16          8 MB
static constexpr size_t OFF_WTF  = OFF_XBF + (size_t)B_*T_*H_*2;        // Wf^T bf16 [2048][512]
static constexpr size_t OFF_UTF  = OFF_WTF + (size_t)H_*G4H*2;
static constexpr size_t OFF_WTB  = OFF_UTF + (size_t)H_*G4H*2;
static constexpr size_t OFF_UTB  = OFF_WTB + (size_t)H_*G4H*2;
static constexpr size_t OFF_XGF  = OFF_UTB + (size_t)H_*G4H*2;          // xg fwd bf16, fragment order [t][ntile][lane][8]
static constexpr size_t OFF_XGB  = OFF_XGF + (size_t)T_*B_*G4H*2;       // xg bwd bf16
static constexpr size_t OFF_HSF  = OFF_XGB + (size_t)T_*B_*G4H*2;       // h seq fwd f32 [B][T][H] 16 MB
static constexpr size_t OFF_HSB  = OFF_HSF + (size_t)B_*T_*H_*4;        // h seq bwd f32           16 MB
static constexpr size_t OFF_HBUF = OFF_HSB + (size_t)B_*T_*H_*4;        // h state [2 dir][2 par][16*512] bf16
static constexpr size_t OFF_CNT  = OFF_HBUF + (size_t)2*2*B_*H_*2;      // barrier counters [2 dir][512] int

// ---------------- init: zero h state + barrier counters (every launch -> graph safe) ----
__global__ void zero_init(bf16* __restrict__ hbuf, int* __restrict__ cnt) {
  int i = blockIdx.x * blockDim.x + threadIdx.x;
  if (i < 2*2*B_*H_) hbuf[i] = (bf16)0.0f;
  if (i < 2*T_)      cnt[i]  = 0;
}

// ---------------- pack: x -> bf16 ----------------
__global__ void pack_x(const float* __restrict__ x, bf16* __restrict__ xb, int n) {
  int i = blockIdx.x * blockDim.x + threadIdx.x;
  if (i < n) xb[i] = (bf16)x[i];
}

// ---------------- pack: W[512][2048] -> W^T bf16 [2048][512] (K contiguous) ----------------
__global__ void pack_wT(const float* __restrict__ w, bf16* __restrict__ wt) {
  int i = blockIdx.x * blockDim.x + threadIdx.x;   // over 2048*512
  int n = i >> 9, k = i & 511;
  wt[i] = (bf16)w[(size_t)k * G4H + n];
}

// ---------------- input GEMM: xg = x*W + b ----------------
// M-tile = the 16 batches at one timestep (same tiling as the scan), so the
// output can be stored in WMMA C-fragment order: xg[((t*128+ntile)*32+lane)*8+v].
// grid (512 timesteps, 16 ntile-groups, 2 dirs), 256 threads = 8 waves, 16x16 tile/wave.
__global__ __launch_bounds__(256) void input_gemm(
    const bf16* __restrict__ xb,
    const bf16* __restrict__ wtF, const bf16* __restrict__ wtB,
    const float* __restrict__ bF, const float* __restrict__ bB,
    bf16* __restrict__ xgF, bf16* __restrict__ xgB)
{
  const int wave = threadIdx.x >> 5, lane = threadIdx.x & 31;
  const int nl = lane & 15, hi = lane >> 4;
  const int t     = blockIdx.x;              // timestep = M tile
  const int ntile = blockIdx.y * 8 + wave;   // 0..127
  const int dir = blockIdx.z;
  const bf16*  wt   = dir ? wtB : wtF;
  const float* bias = dir ? bB  : bF;
  bf16*        xg   = dir ? xgB : xgF;

  const int n0 = ntile * 16;
  const size_t rowA = ((size_t)nl * T_ + t) * H_;  // A row: batch=lane%16, fixed t
  const int kbA = hi ? 8 : 0;                // A K-halves: {kb..kb+7, kb+16..kb+23}
  const int kbB = hi ? 8 : 0;                // B: lane%16 = N col, K-halves {kb..kb+7, kb+16..kb+23}

  v8f acc = {};
  #pragma unroll
  for (int kc = 0; kc < 16; ++kc) {
    union { v16bf v; struct { v8bf lo, hi; } p; } a, b;
    const bf16* ap = xb + rowA + kbA + 32*kc;
    a.p.lo = *(const v8bf*)(ap);
    a.p.hi = *(const v8bf*)(ap + 16);
    const bf16* bp = wt + (size_t)(n0 + nl) * H_ + kbB + 32*kc;
    b.p.lo = *(const v8bf*)(bp);
    b.p.hi = *(const v8bf*)(bp + 16);
    acc = __builtin_amdgcn_wmma_f32_16x16x32_bf16(false, a.v, false, b.v,
                                                  (short)0, acc, false, false);
  }
  const float bv = bias[n0 + nl];
  v8bf frag;
  #pragma unroll
  for (int v = 0; v < 8; ++v) frag[v] = (bf16)(acc[v] + bv);
  // one contiguous 16B store per lane, in exactly the scan's C-fragment order
  *(v8bf*)(xg + (((size_t)t * NTILES + ntile) * 32 + lane) * 8) = frag;
}

// ---------------- persistent recurrent scan ----------------
// 64 blocks: dir = blk>>5, 32 blocks/dir each owning 16 h-columns.
// 128 threads = 4 waves, wave w = gate w (i,f,c,o). U tile (16 cols x 512 K) lives in VGPRs.
__global__ __launch_bounds__(128) void lstm_scan(
    const bf16* __restrict__ utF, const bf16* __restrict__ utB,
    const bf16* __restrict__ xgF, const bf16* __restrict__ xgB,
    float* __restrict__ hsF, float* __restrict__ hsB,
    bf16* __restrict__ hbuf, int* __restrict__ cnt)
{
  const int dir  = blockIdx.x >> 5;
  const int wgid = blockIdx.x & 31;
  const int wave = threadIdx.x >> 5;       // gate index (i,f,c,o)
  const int lane = threadIdx.x & 31;
  const int nl = lane & 15, hi = lane >> 4;
  const int col0   = wgid * 16;            // h-column base
  const int gcol0  = wave * H_ + col0;     // gate-column base in 4H
  const int ntileg = wave * 32 + wgid;     // this wave's N-tile index in [0,128)
  const bf16* ut = dir ? utB : utF;
  const bf16* xg = dir ? xgB : xgF;
  float*      hs = dir ? hsB : hsF;
  bf16* hb  = hbuf + (size_t)dir * 2 * B_ * H_;
  int*  bar = cnt + dir * T_;

  __shared__ float gbuf[4][16][16];        // [gate][m][j]
  __shared__ float cstate[16][16];         // persistent c for this WG's columns
  for (int i = threadIdx.x; i < 256; i += 128) ((float*)cstate)[i] = 0.0f;

  // U^T slab -> 128 VGPRs (16 fragments), resident for the whole scan
  union F { v16bf v; struct { v8bf lo, hi; } p; };
  F u[16];
  {
    const bf16* bp = ut + (size_t)(gcol0 + nl) * H_ + (hi ? 8 : 0);
    #pragma unroll
    for (int kc = 0; kc < 16; ++kc) {
      u[kc].p.lo = *(const v8bf*)(bp + 32*kc);
      u[kc].p.hi = *(const v8bf*)(bp + 32*kc + 16);
    }
  }
  __syncthreads();

  const int kbA = hi ? 8 : 0;
  const size_t fragStride = (size_t)NTILES * 32 * 8;     // xg elems per timestep
  const bf16* myfrag = xg + ((size_t)ntileg * 32 + lane) * 8;

  for (int t = 0; t < T_; ++t) {
    const bf16* hprev = hb + (t & 1) * (B_ * H_);
    bf16*       hnext = hb + ((t + 1) & 1) * (B_ * H_);
    const int tt = dir ? (T_ - 1 - t) : t;   // bwd consumes reversed input

    // C init: one b128 load per lane (fragment-order xg), bias already folded in
    v8bf cfrag = *(const v8bf*)(myfrag + (size_t)tt * fragStride);
    v8f acc;
    #pragma unroll
    for (int v = 0; v < 8; ++v) acc[v] = (float)cfrag[v];

    // prefetch next step's fragment into cache while the WMMA chain runs
    if (t + 1 < T_) {
      const int tn = dir ? (T_ - 2 - t) : (t + 1);
      __builtin_prefetch(myfrag + (size_t)tn * fragStride, 0, 1);
    }

    #pragma unroll
    for (int kc = 0; kc < 16; ++kc) {        // h_prev @ U : 16 chained WMMAs
      F a;
      const bf16* ap = hprev + (size_t)nl * H_ + kbA + 32*kc;
      a.p.lo = *(const v8bf*)(ap);
      a.p.hi = *(const v8bf*)(ap + 16);
      acc = __builtin_amdgcn_wmma_f32_16x16x32_bf16(false, a.v, false, u[kc].v,
                                                    (short)0, acc, false, false);
    }
    #pragma unroll
    for (int v = 0; v < 8; ++v) gbuf[wave][v + 8*hi][nl] = acc[v];
    __syncthreads();

    // elementwise gate mix: 256 (m,j) cells over 128 threads
    for (int e = threadIdx.x; e < 256; e += 128) {
      int m = e >> 4, j = e & 15;
      float gi = gbuf[0][m][j], gf = gbuf[1][m][j];
      float gc = gbuf[2][m][j], go = gbuf[3][m][j];
      gi = 1.0f / (1.0f + __expf(-gi));
      gf = 1.0f / (1.0f + __expf(-gf));
      go = 1.0f / (1.0f + __expf(-go));
      float cn = gf * cstate[m][j] + gi * tanhf(gc);
      cstate[m][j] = cn;
      float h = go * tanhf(cn);
      hnext[m * H_ + col0 + j] = (bf16)h;                       // publish for next step
      hs[((size_t)m * T_ + t) * H_ + col0 + j] = h;             // sequence output @ t
    }
    __syncthreads();

    // per-direction grid barrier (fresh counter per step; zeroed every launch)
    __threadfence();
    if (threadIdx.x == 0) {
      __hip_atomic_fetch_add(&bar[t], 1, __ATOMIC_ACQ_REL, __HIP_MEMORY_SCOPE_AGENT);
      while (__hip_atomic_load(&bar[t], __ATOMIC_ACQUIRE, __HIP_MEMORY_SCOPE_AGENT) < 32)
        __builtin_amdgcn_s_sleep(1);
    }
    __syncthreads();
  }
}

// ---------------- fused residual add + LayerNorm ----------------
__global__ __launch_bounds__(256) void add_ln(
    const float* __restrict__ x, const float* __restrict__ hsF, const float* __restrict__ hsB,
    const float* __restrict__ gamma, const float* __restrict__ beta, float* __restrict__ out)
{
  const size_t base = (size_t)blockIdx.x * H_;
  const int t0 = threadIdx.x, t1 = threadIdx.x + 256;
  __shared__ float ssum[256], ssq[256];
  float y0 = x[base + t0] + hsF[base + t0] + hsB[base + t0];
  float y1 = x[base + t1] + hsF[base + t1] + hsB[base + t1];
  ssum[threadIdx.x] = y0 + y1;
  ssq[threadIdx.x]  = y0*y0 + y1*y1;
  __syncthreads();
  for (int s = 128; s > 0; s >>= 1) {
    if (threadIdx.x < s) {
      ssum[threadIdx.x] += ssum[threadIdx.x + s];
      ssq[threadIdx.x]  += ssq[threadIdx.x + s];
    }
    __syncthreads();
  }
  float mu  = ssum[0] * (1.0f / H_);
  float var = ssq[0] * (1.0f / H_) - mu * mu;
  float inv = rsqrtf(var + 1e-3f);
  out[base + t0] = (y0 - mu) * inv * gamma[t0] + beta[t0];
  out[base + t1] = (y1 - mu) * inv * gamma[t1] + beta[t1];
}

// ---------------- launch ----------------
extern "C" void kernel_launch(void* const* d_in, const int* in_sizes, int n_in,
                              void* d_out, int out_size, void* d_ws, size_t ws_size,
                              hipStream_t stream) {
  const float* x     = (const float*)d_in[0];
  const float* Wf    = (const float*)d_in[1];
  const float* Uf    = (const float*)d_in[2];
  const float* bf    = (const float*)d_in[3];
  const float* Wb    = (const float*)d_in[4];
  const float* Ub    = (const float*)d_in[5];
  const float* bb    = (const float*)d_in[6];
  const float* gamma = (const float*)d_in[7];
  const float* beta  = (const float*)d_in[8];
  float* out = (float*)d_out;

  char* ws = (char*)d_ws;
  bf16*  xb   = (bf16*)(ws + OFF_XBF);
  bf16*  wtF  = (bf16*)(ws + OFF_WTF);
  bf16*  utF  = (bf16*)(ws + OFF_UTF);
  bf16*  wtB  = (bf16*)(ws + OFF_WTB);
  bf16*  utB  = (bf16*)(ws + OFF_UTB);
  bf16*  xgF  = (bf16*)(ws + OFF_XGF);
  bf16*  xgB  = (bf16*)(ws + OFF_XGB);
  float* hsF  = (float*)(ws + OFF_HSF);
  float* hsB  = (float*)(ws + OFF_HSB);
  bf16*  hbuf = (bf16*)(ws + OFF_HBUF);
  int*   cnt  = (int*) (ws + OFF_CNT);

  zero_init<<<128, 256, 0, stream>>>(hbuf, cnt);
  pack_x   <<<(B_*T_*H_ + 255) / 256, 256, 0, stream>>>(x, xb, B_*T_*H_);
  pack_wT  <<<(H_*G4H) / 256, 256, 0, stream>>>(Wf, wtF);
  pack_wT  <<<(H_*G4H) / 256, 256, 0, stream>>>(Uf, utF);
  pack_wT  <<<(H_*G4H) / 256, 256, 0, stream>>>(Wb, wtB);
  pack_wT  <<<(H_*G4H) / 256, 256, 0, stream>>>(Ub, utB);
  input_gemm<<<dim3(512, 16, 2), 256, 0, stream>>>(xb, wtF, wtB, bf, bb, xgF, xgB);
  lstm_scan<<<64, 128, 0, stream>>>(utF, utB, xgF, xgB, hsF, hsB, hbuf, cnt);
  add_ln<<<B_ * T_, 256, 0, stream>>>(x, hsF, hsB, gamma, beta, out);
}
